// MLAttention_78683800863106
// MI455X (gfx1250) — matrix operand order repaired and checked by reading, not compiled, not run on hardware
//
#include <hip/hip_runtime.h>
#include <hip/hip_bf16.h>
#include <math.h>

// ---------------------------------------------------------------------------
// MLA attention for MI455X (gfx1250, wave32, WMMA).
// GEMMs: bf16 WMMA with double-buffered async global->LDS tile staging
// (global_load_async_to_lds_b128 + s_wait_asynccnt). Attention: flash-style
// WMMA with per-wave LDS transpose of the probability tile.
// ---------------------------------------------------------------------------

typedef __attribute__((ext_vector_type(16))) __bf16 v16bf;
typedef __attribute__((ext_vector_type(8)))  __bf16 v8bf;
typedef __attribute__((ext_vector_type(8)))  float  v8f;

#define D_MODEL   2048
#define N_HEADS   16
#define D_HEAD    128
#define DQ        682
#define DQ_PADK   704     // K padding for c_q @ W_u_q^T (multiple of 32)
#define DQ_PADN   768     // N padding for x @ W_d_q^T  (multiple of 128)
#define D_KV      1024
#define D_CKV     1088    // D_KV + 64 (k_r)
#define D_CKV_PAD 1152    // N padding (multiple of 128)
#define D_UKV     3072    // H * (dh + dqn) = 16*192
#define T_SEQ     2048
#define BATCH     2
#define MTOT      (BATCH * T_SEQ)  // 4096

// ----------------------------- WMMA helpers --------------------------------

__device__ __forceinline__ v8f wmma_bf16(v16bf a, v16bf b, v8f c) {
  return __builtin_amdgcn_wmma_f32_16x16x32_bf16(false, a, false, b, (short)0, c,
                                                 false, false);
}

// A fragment (16x32 bf16) from row-major [M][K] in global memory.
__device__ __forceinline__ v16bf ldA(const __bf16* __restrict__ base, int ld,
                                     int mbase, int k0, int lane) {
  const __bf16* p =
      base + (size_t)(mbase + (lane & 15)) * ld + k0 + ((lane >> 4) << 3);
  v8bf c0 = *(const v8bf*)(p);
  v8bf c1 = *(const v8bf*)(p + 16);
  v16bf a;
#pragma unroll
  for (int i = 0; i < 8; ++i) { a[i] = c0[i]; a[i + 8] = c1[i]; }
  return a;
}

// A fragment from an LDS row base (row stride = 32 elements).
__device__ __forceinline__ v16bf ldA_lds(const __bf16* rowp, int lane) {
  const __bf16* p = rowp + ((lane >> 4) << 3);
  v8bf c0 = *(const v8bf*)(p);
  v8bf c1 = *(const v8bf*)(p + 16);
  v16bf a;
#pragma unroll
  for (int i = 0; i < 8; ++i) { a[i] = c0[i]; a[i + 8] = c1[i]; }
  return a;
}

// B fragment (32x16 bf16) from BT = [N][K] row-major (i.e. B transposed).
__device__ __forceinline__ v16bf ldB(const __bf16* __restrict__ baseT, int ld,
                                     int nbase, int k0, int lane) {
  const __bf16* p =
      baseT + (size_t)(nbase + (lane & 15)) * ld + k0 + ((lane >> 4) << 3);
  v8bf c0 = *(const v8bf*)(p);
  v8bf c1 = *(const v8bf*)(p + 16);
  v16bf b;
#pragma unroll
  for (int i = 0; i < 8; ++i) { b[i] = c0[i]; b[i + 8] = c1[i]; }
  return b;
}

__device__ __forceinline__ float redmax16(float v) {
#pragma unroll
  for (int m = 1; m <= 8; m <<= 1) v = fmaxf(v, __shfl_xor(v, m, 32));
  return v;
}
__device__ __forceinline__ float redsum16(float v) {
#pragma unroll
  for (int m = 1; m <= 8; m <<= 1) v += __shfl_xor(v, m, 32);
  return v;
}

// 16B async copy: global -> LDS (tracked by ASYNCcnt).
// Low 32 bits of a generic pointer to __shared__ are the LDS byte offset.
__device__ __forceinline__ void async_cp16(const void* ldsp,
                                           const __bf16* __restrict__ g) {
  unsigned off = (unsigned)(unsigned long long)ldsp;
  asm volatile("global_load_async_to_lds_b128 %0, %1, off"
               :: "v"(off), "v"(g) : "memory");
}

// ----------------------- f32 -> bf16 convert with padding ------------------

__global__ void cvt_pad_bf16(const float* __restrict__ src, __bf16* __restrict__ dst,
                             int srows, int scols, int drows, int dcols) {
  int idx = blockIdx.x * blockDim.x + threadIdx.x;
  int total = drows * dcols;
  if (idx >= total) return;
  int r = idx / dcols, c = idx % dcols;
  float v = (r < srows && c < scols) ? src[(size_t)r * scols + c] : 0.0f;
  dst[idx] = (__bf16)v;
}

// ------------------------------- LayerNorm ---------------------------------

__global__ __launch_bounds__(256)
void ln_to_bf16(const float* __restrict__ src, int ldsrc,
                const float* __restrict__ g, const float* __restrict__ b,
                __bf16* __restrict__ dst, int lddst, int nvalid) {
  int row = blockIdx.x;
  int tid = threadIdx.x;
  const float* x = src + (size_t)row * ldsrc;
  float s = 0.f, s2 = 0.f;
  for (int c = tid; c < nvalid; c += 256) {
    float v = x[c];
    s += v; s2 += v * v;
  }
  __shared__ float sh1[256], sh2[256];
  sh1[tid] = s; sh2[tid] = s2;
  __syncthreads();
  for (int off = 128; off > 0; off >>= 1) {
    if (tid < off) { sh1[tid] += sh1[tid + off]; sh2[tid] += sh2[tid + off]; }
    __syncthreads();
  }
  float mu = sh1[0] / (float)nvalid;
  float var = sh2[0] / (float)nvalid - mu * mu;
  float rstd = rsqrtf(var + 1e-5f);
  __bf16* d = dst + (size_t)row * lddst;
  for (int c = tid; c < lddst; c += 256) {
    float v = (c < nvalid) ? (x[c] - mu) * rstd * g[c] + b[c] : 0.0f;
    d[c] = (__bf16)v;
  }
}

// ------------------------------ bf16 NT GEMM -------------------------------
// C[M x ldc](f32) = A[M x K](bf16) * BT[N x K](bf16)^T + bias
// grid = (Npad/128, M/64), block = 256 (8 waves). Wave tile = 32x32.
// A (64x32) and B (128x32) K-step tiles are staged in LDS with
// double-buffered global_load_async_to_lds_b128 (3 x 16B per thread/step).

__global__ __launch_bounds__(256)
void gemm_bf16_nt(const __bf16* __restrict__ A, const __bf16* __restrict__ BT,
                  const float* __restrict__ bias, float* __restrict__ C,
                  int K, int ldc, int Nvalid) {
  __shared__ __align__(32) __bf16 bufA[2][64][32];    // 8 KB
  __shared__ __align__(32) __bf16 bufB[2][128][32];   // 16 KB

  const int tid = threadIdx.x;
  const int lane = tid & 31;
  const int w = tid >> 5;
  const int mloc = (w & 1) * 32;
  const int nloc = (w >> 1) * 32;
  const int mblk = blockIdx.y * 64;
  const int nblk = blockIdx.x * 128;

  auto stage = [&](int buf, int kk) {
    // B tile: 128 rows x 64B = 512 chunks of 16B
#pragma unroll
    for (int j = 0; j < 2; ++j) {
      int c = j * 256 + tid;
      int n = c >> 2, sub = (c & 3) * 8;
      async_cp16(&bufB[buf][n][sub], BT + (size_t)(nblk + n) * K + kk + sub);
    }
    // A tile: 64 rows x 64B = 256 chunks of 16B
    {
      int m = tid >> 2, sub = (tid & 3) * 8;
      async_cp16(&bufA[buf][m][sub], A + (size_t)(mblk + m) * K + kk + sub);
    }
  };

  const int nk = K >> 5;
  stage(0, 0);

  v8f acc00 = {}, acc01 = {}, acc10 = {}, acc11 = {};
  for (int it = 0; it < nk; ++it) {
    asm volatile("s_wait_asynccnt 0x0" ::: "memory");  // own stage-it data landed
    __syncthreads();                                   // everyone's landed
    if (it + 1 < nk) stage((it + 1) & 1, (it + 1) << 5);
    const int cur = it & 1;
    v16bf a0 = ldA_lds(&bufA[cur][mloc + (lane & 15)][0], lane);
    v16bf a1 = ldA_lds(&bufA[cur][mloc + 16 + (lane & 15)][0], lane);
    v16bf b0 = ldA_lds(&bufB[cur][nloc + (lane & 15)][0], lane);
    v16bf b1 = ldA_lds(&bufB[cur][nloc + 16 + (lane & 15)][0], lane);
    acc00 = wmma_bf16(a0, b0, acc00);
    acc01 = wmma_bf16(a0, b1, acc01);
    acc10 = wmma_bf16(a1, b0, acc10);
    acc11 = wmma_bf16(a1, b1, acc11);
  }

  const int m0 = mblk + mloc;
  const int n0 = nblk + nloc;
  auto st = [&](v8f acc, int mb, int nb) {
    int col = nb + (lane & 15);
    float bv = (col < Nvalid) ? bias[col] : 0.0f;
#pragma unroll
    for (int r = 0; r < 8; ++r) {
      int row = mb + r + ((lane >> 4) << 3);
      C[(size_t)row * ldc + col] = acc[r] + bv;
    }
  };
  st(acc00, m0, n0);
  st(acc01, m0, n0 + 16);
  st(acc10, m0 + 16, n0);
  st(acc11, m0 + 16, n0 + 16);
}

// ------------------------- assemble q/k/v (rotary) -------------------------

__device__ __forceinline__ float inv_freq32(int j) {
  // (1/10000)^(j/32) = exp(-j * ln(10000)/32)
  return __expf(-(float)j * 0.28782313662425575f);
}

__global__ void assemble_q(const float* __restrict__ qlin, __bf16* __restrict__ qf) {
  int idx = blockIdx.x * blockDim.x + threadIdx.x;  // MTOT*16*32
  if (idx >= MTOT * N_HEADS * 32) return;
  int j = idx & 31;
  int h = (idx >> 5) & 15;
  int bt = idx >> 9;
  int t = bt & (T_SEQ - 1);
  int b = bt >> 11;
  const float* q = qlin + (size_t)bt * D_MODEL + h * D_HEAD;
  float ang = (float)t * inv_freq32(j);
  float cs = __cosf(ang), sn = __sinf(ang);
  float x1 = q[64 + j], x2 = q[96 + j];
  size_t ob = ((size_t)(b * N_HEADS + h) * T_SEQ + t) * D_HEAD;
  qf[ob + j]      = (__bf16)q[j];
  qf[ob + 32 + j] = (__bf16)q[32 + j];
  qf[ob + 64 + j] = (__bf16)(x1 * cs + x2 * sn);
  qf[ob + 96 + j] = (__bf16)(-x1 * sn + x2 * cs);
}

__global__ void assemble_k(const float* __restrict__ kvlin,
                           const float* __restrict__ ckvlin,
                           __bf16* __restrict__ kf) {
  int idx = blockIdx.x * blockDim.x + threadIdx.x;  // MTOT*16*32
  if (idx >= MTOT * N_HEADS * 32) return;
  int j = idx & 31;
  int h = (idx >> 5) & 15;
  int bt = idx >> 9;
  int t = bt & (T_SEQ - 1);
  int b = bt >> 11;
  const float* kn = kvlin + (size_t)bt * D_UKV + h * 192;
  const float* kr = ckvlin + (size_t)bt * D_CKV_PAD + D_KV;
  float ang = (float)t * inv_freq32(j);
  float cs = __cosf(ang), sn = __sinf(ang);
  float x1 = kr[j], x2 = kr[32 + j];
  size_t ob = ((size_t)(b * N_HEADS + h) * T_SEQ + t) * D_HEAD;
  kf[ob + j]      = (__bf16)kn[j];
  kf[ob + 32 + j] = (__bf16)kn[32 + j];
  kf[ob + 64 + j] = (__bf16)(x1 * cs + x2 * sn);
  kf[ob + 96 + j] = (__bf16)(-x1 * sn + x2 * cs);
}

// V^T: vt[bh][d][t] = kv[bt][h*192 + 64 + d]
__global__ void assemble_vt(const float* __restrict__ kvlin, __bf16* __restrict__ vt) {
  int idx = blockIdx.x * blockDim.x + threadIdx.x;  // MTOT*16*128
  if (idx >= MTOT * N_HEADS * D_HEAD) return;
  int d = idx & 127;
  int h = (idx >> 7) & 15;
  int bt = idx >> 11;
  int t = bt & (T_SEQ - 1);
  int b = bt >> 11;
  vt[((size_t)(b * N_HEADS + h) * D_HEAD + d) * T_SEQ + t] =
      (__bf16)kvlin[(size_t)bt * D_UKV + h * 192 + 64 + d];
}

// ------------------------- flash attention (WMMA) --------------------------
// grid = (T/128, B*H), block = 256. Each wave owns 16 query rows.

__global__ __launch_bounds__(256)
void attn_flash(const __bf16* __restrict__ qf, const __bf16* __restrict__ kfull,
                const __bf16* __restrict__ vt, __bf16* __restrict__ ybf) {
  __shared__ __align__(32) __bf16 smem[8][16][32];  // per-wave P transpose buffer

  const int lane = threadIdx.x & 31;
  const int w = threadIdx.x >> 5;
  const int bh = blockIdx.y;
  const int b = bh >> 4, h = bh & 15;
  const int m0 = (blockIdx.x * 8 + w) * 16;  // query tile origin in [0, T)
  const int hi = lane >> 4;
  const int c0 = lane & 15;

  const __bf16* Q  = qf    + (size_t)bh * T_SEQ * D_HEAD;
  const __bf16* Kf = kfull + (size_t)bh * T_SEQ * D_HEAD;
  const __bf16* VT = vt    + (size_t)bh * D_HEAD * T_SEQ;

  // Q A-fragments for the 4 d-chunks (d = 0..127)
  v16bf qa[4];
#pragma unroll
  for (int j = 0; j < 4; ++j) qa[j] = ldA(Q, D_HEAD, m0, j * 32, lane);

  v8f acc[8] = {};
  float mrow[8], lrow[8];
#pragma unroll
  for (int r = 0; r < 8; ++r) { mrow[r] = -1e30f; lrow[r] = 0.0f; }
  const float scale = 0.08838834764831845f;  // 1/sqrt(128)

  const int kend = m0 + 16;  // causal: only kpos <= m0+15 needed
  for (int k0 = 0; k0 < kend; k0 += 32) {
    // ---- scores: S = Q * K^T for kpos tiles [k0,k0+16) and [k0+16,k0+32)
    v8f s0 = {}, s1 = {};
#pragma unroll
    for (int j = 0; j < 4; ++j) {
      v16bf bk0 = ldB(Kf, D_HEAD, k0, j * 32, lane);
      v16bf bk1 = ldB(Kf, D_HEAD, k0 + 16, j * 32, lane);
      s0 = wmma_bf16(qa[j], bk0, s0);
      s1 = wmma_bf16(qa[j], bk1, s1);
    }

    // ---- scale + causal mask + online softmax
    float p0v[8], p1v[8], fac[8];
#pragma unroll
    for (int r = 0; r < 8; ++r) {
      int rowq = m0 + r + (hi << 3);
      float a0 = (k0 + c0 <= rowq) ? s0[r] * scale : -1e30f;
      float a1 = (k0 + 16 + c0 <= rowq) ? s1[r] * scale : -1e30f;
      float tmax = redmax16(fmaxf(a0, a1));
      float mnew = fmaxf(mrow[r], tmax);
      float f = __expf(mrow[r] - mnew);
      mrow[r] = mnew;
      float e0 = __expf(a0 - mnew);
      float e1 = __expf(a1 - mnew);
      lrow[r] = lrow[r] * f + redsum16(e0 + e1);
      fac[r] = f;
      p0v[r] = e0;
      p1v[r] = e1;
    }
#pragma unroll
    for (int j = 0; j < 8; ++j)
#pragma unroll
      for (int r = 0; r < 8; ++r) acc[j][r] = acc[j][r] * fac[r];

    // ---- transpose P (D-layout -> A-layout) through per-wave LDS slice
#pragma unroll
    for (int r = 0; r < 8; ++r) {
      smem[w][r + (hi << 3)][c0]      = (__bf16)p0v[r];
      smem[w][r + (hi << 3)][c0 + 16] = (__bf16)p1v[r];
    }
    asm volatile("s_wait_dscnt 0" ::: "memory");
    v16bf pa;
    {
      const __bf16* lp = &smem[w][c0][hi << 3];
      v8bf pc0 = *(const v8bf*)(lp);
      v8bf pc1 = *(const v8bf*)(lp + 16);
#pragma unroll
      for (int i = 0; i < 8; ++i) { pa[i] = pc0[i]; pa[i + 8] = pc1[i]; }
    }

    // ---- acc += P * V  (B fragments contiguous from V^T [d][t])
#pragma unroll
    for (int j = 0; j < 8; ++j) {
      v16bf bv = ldB(VT, T_SEQ, j * 16, k0, lane);
      acc[j] = wmma_bf16(pa, bv, acc[j]);
    }
  }

  // ---- normalize and emit y as bf16 [B*T, 2048] (d-major per head)
#pragma unroll
  for (int r = 0; r < 8; ++r) {
    float inv = 1.0f / lrow[r];
    int t = m0 + r + (hi << 3);
    size_t rowbt = (size_t)b * T_SEQ + t;
#pragma unroll
    for (int j = 0; j < 8; ++j) {
      ybf[rowbt * D_MODEL + h * D_HEAD + j * 16 + c0] = (__bf16)(acc[j][r] * inv);
    }
  }
}

// ------------------------------- launcher ----------------------------------

extern "C" void kernel_launch(void* const* d_in, const int* in_sizes, int n_in,
                              void* d_out, int out_size, void* d_ws, size_t ws_size,
                              hipStream_t stream) {
  const float* x     = (const float*)d_in[0];
  const float* Wdq   = (const float*)d_in[1];
  const float* bdq   = (const float*)d_in[2];
  const float* qg    = (const float*)d_in[3];
  const float* qb    = (const float*)d_in[4];
  const float* Wuq   = (const float*)d_in[5];
  const float* buq   = (const float*)d_in[6];
  const float* Wdkv  = (const float*)d_in[7];
  const float* bdkv  = (const float*)d_in[8];
  const float* kvg   = (const float*)d_in[9];
  const float* kvb   = (const float*)d_in[10];
  const float* Wukv  = (const float*)d_in[11];
  const float* bukv  = (const float*)d_in[12];
  const float* Wo    = (const float*)d_in[13];
  const float* bo    = (const float*)d_in[14];
  float* out = (float*)d_out;

  // workspace carve-up (256B aligned)
  char* p = (char*)d_ws;
  auto alloc = [&](size_t bytes) -> void* {
    void* r = (void*)p;
    p += (bytes + 255) & ~(size_t)255;
    return r;
  };
  __bf16* x_bf    = (__bf16*)alloc((size_t)MTOT * D_MODEL * 2);
  __bf16* wdq_bf  = (__bf16*)alloc((size_t)DQ_PADN * D_MODEL * 2);
  __bf16* wuq_bf  = (__bf16*)alloc((size_t)D_MODEL * DQ_PADK * 2);
  __bf16* wdkv_bf = (__bf16*)alloc((size_t)D_CKV_PAD * D_MODEL * 2);
  __bf16* wukv_bf = (__bf16*)alloc((size_t)D_UKV * D_KV * 2);
  __bf16* wo_bf   = (__bf16*)alloc((size_t)D_MODEL * D_MODEL * 2);
  float*  cq_lin  = (float*)alloc((size_t)MTOT * DQ_PADN * 4);
  __bf16* cq_bf   = (__bf16*)alloc((size_t)MTOT * DQ_PADK * 2);
  float*  q_lin   = (float*)alloc((size_t)MTOT * D_MODEL * 4);
  float*  ckv_lin = (float*)alloc((size_t)MTOT * D_CKV_PAD * 4);
  __bf16* kvn_bf  = (__bf16*)alloc((size_t)MTOT * D_KV * 2);
  float*  kv_lin  = (float*)alloc((size_t)MTOT * D_UKV * 4);
  __bf16* qf      = (__bf16*)alloc((size_t)BATCH * N_HEADS * T_SEQ * D_HEAD * 2);
  __bf16* kfull   = (__bf16*)alloc((size_t)BATCH * N_HEADS * T_SEQ * D_HEAD * 2);
  __bf16* vt      = (__bf16*)alloc((size_t)BATCH * N_HEADS * D_HEAD * T_SEQ * 2);
  __bf16* ybf     = (__bf16*)alloc((size_t)MTOT * D_MODEL * 2);

  auto cvt = [&](const float* s, __bf16* d, int sr, int sc, int dr, int dc) {
    int total = dr * dc;
    cvt_pad_bf16<<<(total + 255) / 256, 256, 0, stream>>>(s, d, sr, sc, dr, dc);
  };

  // 1) convert x + weights to (padded) bf16
  cvt(x, x_bf, MTOT, D_MODEL, MTOT, D_MODEL);
  cvt(Wdq, wdq_bf, DQ, D_MODEL, DQ_PADN, D_MODEL);
  cvt(Wuq, wuq_bf, D_MODEL, DQ, D_MODEL, DQ_PADK);
  cvt(Wdkv, wdkv_bf, D_CKV, D_MODEL, D_CKV_PAD, D_MODEL);
  cvt(Wukv, wukv_bf, D_UKV, D_KV, D_UKV, D_KV);
  cvt(Wo, wo_bf, D_MODEL, D_MODEL, D_MODEL, D_MODEL);

  // 2) c_q = x @ Wdq^T + b  (M=4096, N=768(pad), K=2048)
  gemm_bf16_nt<<<dim3(DQ_PADN / 128, MTOT / 64), 256, 0, stream>>>(
      x_bf, wdq_bf, bdq, cq_lin, D_MODEL, DQ_PADN, DQ);
  // 3) layernorm(c_q) -> bf16, K-padded to 704
  ln_to_bf16<<<MTOT, 256, 0, stream>>>(cq_lin, DQ_PADN, qg, qb, cq_bf, DQ_PADK, DQ);
  // 4) q = c_q @ Wuq^T + b  (N=2048, K=704)
  gemm_bf16_nt<<<dim3(D_MODEL / 128, MTOT / 64), 256, 0, stream>>>(
      cq_bf, wuq_bf, buq, q_lin, DQ_PADK, D_MODEL, D_MODEL);
  // 5) c_kv = x @ Wdkv^T + b  (N=1152(pad), K=2048)
  gemm_bf16_nt<<<dim3(D_CKV_PAD / 128, MTOT / 64), 256, 0, stream>>>(
      x_bf, wdkv_bf, bdkv, ckv_lin, D_MODEL, D_CKV_PAD, D_CKV);
  // 6) layernorm(c_kv[:, :1024]) -> bf16
  ln_to_bf16<<<MTOT, 256, 0, stream>>>(ckv_lin, D_CKV_PAD, kvg, kvb, kvn_bf, D_KV, D_KV);
  // 7) kv = kv_n @ Wukv^T + b  (N=3072, K=1024)
  gemm_bf16_nt<<<dim3(D_UKV / 128, MTOT / 64), 256, 0, stream>>>(
      kvn_bf, wukv_bf, bukv, kv_lin, D_KV, D_UKV, D_UKV);

  // 8) assemble q_full / k_full (rotary) and V^T, all bf16
  {
    int tq = MTOT * N_HEADS * 32;
    assemble_q<<<(tq + 255) / 256, 256, 0, stream>>>(q_lin, qf);
    assemble_k<<<(tq + 255) / 256, 256, 0, stream>>>(kv_lin, ckv_lin, kfull);
    int tv = MTOT * N_HEADS * D_HEAD;
    assemble_vt<<<(tv + 255) / 256, 256, 0, stream>>>(kv_lin, vt);
  }

  // 9) causal flash attention -> y (bf16, [B*T, 2048])
  attn_flash<<<dim3(T_SEQ / 128, BATCH * N_HEADS), 256, 0, stream>>>(qf, kfull, vt, ybf);

  // 10) out = y @ Wo^T + b_o  (f32 output)
  gemm_bf16_nt<<<dim3(D_MODEL / 128, MTOT / 64), 256, 0, stream>>>(
      ybf, wo_bf, bo, out, D_MODEL, D_MODEL, D_MODEL);
}
